// EINModel_51668456571562
// MI455X (gfx1250) — matrix operand order repaired and checked by reading, not compiled, not run on hardware
//
#include <hip/hip_runtime.h>
#include <hip/hip_fp16.h>
#include <math.h>

typedef __attribute__((ext_vector_type(16))) _Float16 v16h;
typedef __attribute__((ext_vector_type(8)))  _Float16 v8h;
typedef __attribute__((ext_vector_type(8)))  float    v8f;

#define TB 256

// ---------------------------------------------------------------- utilities
__global__ void fill_f32(float* __restrict__ p, float v, int n) {
    int i = blockIdx.x * blockDim.x + threadIdx.x;
    if (i < n) p[i] = v;
}

__global__ void cvt_f32_to_f16(const float* __restrict__ src, _Float16* __restrict__ dst, int n) {
    int i = blockIdx.x * blockDim.x + threadIdx.x;
    if (i < n) dst[i] = (_Float16)src[i];
}

// Pack [Wq|Wk|Wv|Wskip] (each [L][64][64] f32, K x Ncol row-major) into a
// TRANSPOSED f16 tensor WcatT [L][256][64] (col-major weights: WcatT[l][j][k]).
// This makes every lane's WMMA B-fragment a contiguous 32-byte chunk.
__global__ void prep_wcatT(const float* __restrict__ Wq, const float* __restrict__ Wk,
                           const float* __restrict__ Wv, const float* __restrict__ Ws,
                           _Float16* __restrict__ WcatT, int Lc) {
    int i = blockIdx.x * blockDim.x + threadIdx.x;
    int total = Lc * 256 * 64;
    if (i >= total) return;
    int k = i & 63;           // K row of the weight
    int j = (i >> 6) & 255;   // output column 0..255
    int l = i >> 14;          // layer
    int blk = j >> 6;         // which weight matrix
    int c = j & 63;
    const float* W = (blk == 0) ? Wq : (blk == 1) ? Wk : (blk == 2) ? Wv : Ws;
    WcatT[i] = (_Float16)W[(((size_t)l * 64) + k) * 64 + c];
}

__global__ void count_nodes(const int* __restrict__ batch, float* __restrict__ cnt, int n) {
    int i = blockIdx.x * blockDim.x + threadIdx.x;
    if (i < n) atomicAdd(cnt + batch[i], 1.0f);
}

__global__ void inv_counts(const float* __restrict__ cnt, float* __restrict__ inv, int g) {
    int i = blockIdx.x * blockDim.x + threadIdx.x;
    if (i < g) inv[i] = 1.0f / fmaxf(cnt[i], 1.0f);
}

__device__ inline void atomicMaxF(float* addr, float val) {
    int* ai = (int*)addr;
    int old = __float_as_int(*addr);
    while (__int_as_float(old) < val) {
        int assumed = old;
        old = atomicCAS(ai, assumed, __float_as_int(val));
        if (old == assumed) break;
    }
}

// --------------------------------------------- WMMA node projection kernel
// P[N][256] = hf16[N][64] @ Wcat[64][256]   (f16 in, f32 accum)
// One wave per 16x16 output tile; 16 waves (512 thr) per block cover the 16
// column tiles of one 16-row strip.  K=64 -> two chained v_wmma_f32_16x16x32_f16.
__global__ void node_proj_wmma(const _Float16* __restrict__ hf16,
                               const _Float16* __restrict__ WcatT, // [256][64]
                               float* __restrict__ P, int N) {
    const int wave = threadIdx.x >> 5;    // 0..15 -> column tile
    const int lane = threadIdx.x & 31;
    const int row0 = blockIdx.x * 16;
    const int col0 = wave * 16;

    const int m  = lane & 15;
    const int hi = lane >> 4;

    // A fragment (16-bit A 16x32): lane group 0 holds K=0..15,
    // group 1 holds K=16..31 of row M=lane%16 (matches B's k order).
    int arow = row0 + m;
    if (arow >= N) arow = N - 1;                 // clamp (stores are guarded)
    const _Float16* ap = hf16 + (size_t)arow * 64;
    const int k0 = hi * 16;
    v8h a0l = *(const v8h*)(ap + k0);
    v8h a0h = *(const v8h*)(ap + k0 + 8);
    v8h a1l = *(const v8h*)(ap + 32 + k0);
    v8h a1h = *(const v8h*)(ap + 32 + k0 + 8);
    v16h a0, a1;
#pragma unroll
    for (int j = 0; j < 8; ++j) {
        a0[j] = a0l[j]; a0[j + 8] = a0h[j];
        a1[j] = a1l[j]; a1[j + 8] = a1h[j];
    }

    // B fragment (16-bit B 32x16): lane group 0 -> K=0..15, group 1 -> K=16..31
    // for column N=lane%16.  With transposed weights this is one contiguous
    // 32-byte chunk per fragment.
    const int kb = hi * 16;
    const int col = col0 + m;
    const _Float16* wp = WcatT + (size_t)col * 64;
    v16h b0 = *(const v16h*)(wp + kb);
    v16h b1 = *(const v16h*)(wp + 32 + kb);

    v8f c = {};
    c = __builtin_amdgcn_wmma_f32_16x16x32_f16(false, a0, false, b0, (short)0, c, false, false);
    c = __builtin_amdgcn_wmma_f32_16x16x32_f16(false, a1, false, b1, (short)0, c, false, false);

    // C/D layout: VGPR r -> M = r + 8*(lane>=16), N = lane%16
    float* pout = P + (size_t)(row0 + 8 * hi) * 256 + col;
    if (row0 + 16 <= N) {   // fast path: straight-line stores, no EXEC juggling
#pragma unroll
        for (int r = 0; r < 8; ++r) pout[(size_t)r * 256] = c[r];
    } else {
#pragma unroll
        for (int r = 0; r < 8; ++r)
            if (row0 + 8 * hi + r < N) pout[(size_t)r * 256] = c[r];
    }
}

// -------------------------------------------------- edge pass 1: logits+max
// one thread per (edge, head).  P columns: [0..63]=Q, [64..127]=K,
// [128..191]=V, [192..255]=Skip.
__global__ void edge_logits(const float* __restrict__ P, const float* __restrict__ ea,
                            const float* __restrict__ We,   // [16][64] (this layer)
                            const int* __restrict__ src, const int* __restrict__ dst,
                            float* __restrict__ logits, float* __restrict__ mmax, int E) {
    __shared__ float sWe[16 * 64];
    for (int i = threadIdx.x; i < 16 * 64; i += blockDim.x) sWe[i] = We[i];
    __syncthreads();
    int t = blockIdx.x * blockDim.x + threadIdx.x;
    if (t >= E * 4) return;
    int e = t >> 2, h = t & 3;
    int sN = src[e], dN = dst[e];
    const float* eap = ea + (size_t)e * 16;
    const float* qp = P + (size_t)dN * 256 + h * 16;
    const float* kp = P + (size_t)sN * 256 + 64 + h * 16;
    float logit = 0.f;
#pragma unroll
    for (int d = 0; d < 16; ++d) {
        float eh = 0.f;
#pragma unroll
        for (int j = 0; j < 16; ++j) eh += eap[j] * sWe[j * 64 + h * 16 + d];
        logit += qp[d] * (kp[d] + eh);
    }
    logit *= 0.25f;                 // SCALE = 1/sqrt(16)
    logits[t] = logit;
    atomicMaxF(mmax + (size_t)dN * 4 + h, logit);
}

// ----------------------------------- edge pass 2: exp, sum, weighted V agg
__global__ void edge_accum(const float* __restrict__ P, const float* __restrict__ ea,
                           const float* __restrict__ We,
                           const int* __restrict__ src, const int* __restrict__ dst,
                           const float* __restrict__ logits, const float* __restrict__ mmax,
                           float* __restrict__ ssum, float* __restrict__ agg, int E) {
    __shared__ float sWe[16 * 64];
    for (int i = threadIdx.x; i < 16 * 64; i += blockDim.x) sWe[i] = We[i];
    __syncthreads();
    int t = blockIdx.x * blockDim.x + threadIdx.x;
    if (t >= E * 4) return;
    int e = t >> 2, h = t & 3;
    int sN = src[e], dN = dst[e];
    const float* eap = ea + (size_t)e * 16;
    const float* vp = P + (size_t)sN * 256 + 128 + h * 16;
    float a = expf(logits[t] - mmax[(size_t)dN * 4 + h]);
    atomicAdd(ssum + (size_t)dN * 4 + h, a);
    float* ag = agg + (size_t)dN * 64 + h * 16;
#pragma unroll
    for (int d = 0; d < 16; ++d) {
        float eh = 0.f;
#pragma unroll
        for (int j = 0; j < 16; ++j) eh += eap[j] * sWe[j * 64 + h * 16 + d];
        atomicAdd(ag + d, a * (vp[d] + eh));
    }
}

// --------------------------- node finalize: normalize + skip + relu + pool
// Also emits the f16 copy of h so the next layer's WMMA kernel can consume it
// without a separate conversion pass.
__global__ void node_finalize(const float* __restrict__ P, const float* __restrict__ ssum,
                              const float* __restrict__ agg, const int* __restrict__ batch,
                              const float* __restrict__ invcnt,
                              float* __restrict__ h, _Float16* __restrict__ hf,
                              float* __restrict__ pooled_l, int N) {
    int t = blockIdx.x * blockDim.x + threadIdx.x;
    if (t >= N * 64) return;
    int n = t >> 6, d = t & 63, hh = d >> 4;
    float v = agg[t] / (ssum[(size_t)n * 4 + hh] + 1e-16f) + P[(size_t)n * 256 + 192 + d];
    v = fmaxf(v, 0.f);
    h[t] = v;
    hf[t] = (_Float16)v;
    int g = batch[n];
    atomicAdd(pooled_l + (size_t)g * 64 + d, v * invcnt[g]);
}

// --------------------------------------------------------------- final MLP
// pooled: [L][G][64]; hcat column index i = l*64+d.  One block per graph.
__global__ void mlp_head(const float* __restrict__ pooled, const float* __restrict__ W1,
                         const float* __restrict__ b1, const float* __restrict__ W2,
                         const float* __restrict__ b2, float* __restrict__ out,
                         int Lc, int G) {
    int g = blockIdx.x;
    int j = threadIdx.x;            // 0..Lc*64-1 (=320)
    __shared__ float acc;
    float s = b1[j];
    for (int l = 0; l < Lc; ++l) {
        const float* pr = pooled + ((size_t)l * G + g) * 64;
        for (int d = 0; d < 64; ++d)
            s += pr[d] * W1[(size_t)(l * 64 + d) * (Lc * 64) + j];
    }
    float r = fmaxf(s, 0.f);
    if (j == 0) acc = 0.f;
    __syncthreads();
    atomicAdd(&acc, r * W2[j]);
    __syncthreads();
    if (j == 0) out[g] = acc + b2[0];
}

// ------------------------------------------------------------------ driver
extern "C" void kernel_launch(void* const* d_in, const int* in_sizes, int n_in,
                              void* d_out, int out_size, void* d_ws, size_t ws_size,
                              hipStream_t stream) {
    (void)n_in; (void)ws_size;
    const float* x    = (const float*)d_in[0];
    const int*   eidx = (const int*)d_in[1];
    const float* ea   = (const float*)d_in[2];
    const int*   bat  = (const int*)d_in[3];
    const float* Wq   = (const float*)d_in[4];
    const float* Wk   = (const float*)d_in[5];
    const float* Wv   = (const float*)d_in[6];
    const float* We   = (const float*)d_in[7];
    const float* Ws   = (const float*)d_in[8];
    const float* W1   = (const float*)d_in[9];
    const float* b1   = (const float*)d_in[10];
    const float* W2   = (const float*)d_in[11];
    const float* b2   = (const float*)d_in[12];

    const int N  = in_sizes[0] / 64;
    const int E  = in_sizes[1] / 2;
    const int Lc = in_sizes[4] / (64 * 64);
    const int G  = out_size;                 // FINAL == 1
    const int* srcI = eidx;
    const int* dstI = eidx + E;

    char* wsp = (char*)d_ws;
    auto alloc = [&](size_t bytes) -> char* {
        char* p = wsp;
        wsp += (bytes + 255) & ~(size_t)255;
        return p;
    };
    float*    h      = (float*)   alloc((size_t)N * 64 * 4);
    _Float16* hf     = (_Float16*)alloc((size_t)N * 64 * 2);
    float*    P      = (float*)   alloc((size_t)N * 256 * 4);
    float*    logits = (float*)   alloc((size_t)E * 4 * 4);
    float*    mmax   = (float*)   alloc((size_t)N * 4 * 4);
    float*    ssum   = (float*)   alloc((size_t)N * 4 * 4);
    float*    agg    = (float*)   alloc((size_t)N * 64 * 4);
    float*    pooled = (float*)   alloc((size_t)Lc * G * 64 * 4);
    float*    cnt    = (float*)   alloc((size_t)G * 4);
    float*    inv    = (float*)   alloc((size_t)G * 4);
    _Float16* WcatT  = (_Float16*)alloc((size_t)Lc * 256 * 64 * 2);

    // one-time (per call) prep
    prep_wcatT<<<(Lc * 256 * 64 + TB - 1) / TB, TB, 0, stream>>>(Wq, Wk, Wv, Ws, WcatT, Lc);
    fill_f32<<<(G + TB - 1) / TB, TB, 0, stream>>>(cnt, 0.f, G);
    count_nodes<<<(N + TB - 1) / TB, TB, 0, stream>>>(bat, cnt, N);
    inv_counts<<<(G + TB - 1) / TB, TB, 0, stream>>>(cnt, inv, G);
    fill_f32<<<(Lc * G * 64 + TB - 1) / TB, TB, 0, stream>>>(pooled, 0.f, Lc * G * 64);
    cvt_f32_to_f16<<<(N * 64 + TB - 1) / TB, TB, 0, stream>>>(x, hf, N * 64);

    const int rowTiles = (N + 15) / 16;
    for (int l = 0; l < Lc; ++l) {
        node_proj_wmma<<<rowTiles, 512, 0, stream>>>(hf, WcatT + (size_t)l * 256 * 64, P, N);

        fill_f32<<<(N * 4 + TB - 1) / TB, TB, 0, stream>>>(mmax, -3.0e38f, N * 4);
        fill_f32<<<(N * 4 + TB - 1) / TB, TB, 0, stream>>>(ssum, 0.f, N * 4);
        fill_f32<<<(N * 64 + TB - 1) / TB, TB, 0, stream>>>(agg, 0.f, N * 64);

        const float* WeL = We + (size_t)l * 16 * 64;
        int egrid = (E * 4 + TB - 1) / TB;
        edge_logits<<<egrid, TB, 0, stream>>>(P, ea, WeL, srcI, dstI, logits, mmax, E);
        edge_accum <<<egrid, TB, 0, stream>>>(P, ea, WeL, srcI, dstI, logits, mmax, ssum, agg, E);

        node_finalize<<<(N * 64 + TB - 1) / TB, TB, 0, stream>>>(
            P, ssum, agg, bat, inv, h, hf, pooled + (size_t)l * G * 64, N);
    }

    mlp_head<<<G, Lc * 64, 0, stream>>>(pooled, W1, b1, W2, b2, (float*)d_out, Lc, G);
}
